// MLNN_34050500722932
// MI455X (gfx1250) — matrix operand order repaired and checked by reading, compiled, not run on hardware
//
#include <hip/hip_runtime.h>
#include <hip/hip_bf16.h>
#include <stdint.h>

#define B_SZ 16384
#define DIN  1024
#define HDIM 512
#define DOUTDIM 256
#define NEXP 8
#define NMIDL 2
#define KT   64          // K-tile per LDS stage

typedef __attribute__((ext_vector_type(16))) __bf16 bf16x16;
typedef __attribute__((ext_vector_type(8)))  float  f32x8;
typedef int v4i_vs __attribute__((vector_size(16)));

#if __has_builtin(__builtin_amdgcn_global_load_async_to_lds_b128) && \
    __has_builtin(__builtin_amdgcn_s_wait_asynccnt)
#define HAVE_ASYNC_LDS 1
#else
#define HAVE_ASYNC_LDS 0
#endif

__device__ __forceinline__ unsigned short f2bf_u(float f) {
  union { float f; uint32_t u; } a; a.f = f;
  uint32_t u = a.u;
  uint32_t r = u + 0x7fffu + ((u >> 16) & 1u);   // round-to-nearest-even
  return (unsigned short)(r >> 16);
}
__device__ __forceinline__ float bf2f(unsigned short s) {
  union { uint32_t u; float f; } a; a.u = ((uint32_t)s) << 16;
  return a.f;
}

__device__ __forceinline__ void lds_copy16_async(const unsigned short* g,
                                                 unsigned short* l) {
#if HAVE_ASYNC_LDS
  __builtin_amdgcn_global_load_async_to_lds_b128(
      (__attribute__((address_space(1))) v4i_vs*)(void*)g,
      (__attribute__((address_space(3))) v4i_vs*)(void*)l, 0, 0);
#else
  *(uint4*)l = *(const uint4*)g;
#endif
}
__device__ __forceinline__ void wait_async_lds() {
#if HAVE_ASYNC_LDS
  __builtin_amdgcn_s_wait_asynccnt(0);
#endif
}

// ---------------------------------------------------------------------------
// WMMA GEMM: C = relu(A[bf16, MxK] @ Bt[bf16, NxK]^T + bias)
// Block tile 64x128, 8 waves, each wave 32x32 (2x2 of 16x16x32 bf16 WMMA).
// Double-buffered LDS, KT=64 (8 WMMAs per barrier), async global->LDS copy.
// GATHER: rows of A gathered via idxlist (expert dispatch), results scattered.
// ---------------------------------------------------------------------------
template<bool GATHER>
__global__ __launch_bounds__(256) void gemm_kernel(
    const unsigned short* __restrict__ A,
    const unsigned short* __restrict__ Bt,     // [N,K] row-major (pre-transposed W)
    const float* __restrict__ bias,            // [N]
    float* __restrict__ C,                     // [M,N] fp32
    int M, int N, int K,
    const int* __restrict__ idxlist,
    const int* __restrict__ counts,
    const int* __restrict__ bases,
    int strideB, int strideBias)
{
  __shared__ __align__(16) unsigned short As[2][64][72];
  __shared__ __align__(16) unsigned short Bs[2][128][72];
  __shared__ int RowIdx[64];

  const int rowBase = blockIdx.x * 64;
  if (GATHER) {
    const int e = blockIdx.z;
    const int cnt = counts[e];
    if (rowBase >= cnt) return;
    Bt   += (size_t)e * strideB;
    bias += (size_t)e * strideBias;
    const int base = bases[e];
    for (int r = threadIdx.x; r < 64; r += blockDim.x) {
      int g = rowBase + r;
      RowIdx[r] = (g < cnt) ? idxlist[base + g] : -1;
    }
    __syncthreads();
  }
  const int colBase = blockIdx.y * 128;

  const int t    = threadIdx.x;
  const int lane = t & 31;
  const int wave = t >> 5;
  const int waveRow = wave >> 2;   // 0..1
  const int waveCol = wave & 3;    // 0..3

  f32x8 acc00 = {}, acc01 = {}, acc10 = {}, acc11 = {};

  // A loader: thread t owns row t/4, two 16B chunks at k = (t&3)*16, +8
  const int ar  = t >> 2;
  const int akb = (t & 3) * 16;
  size_t aRow;
  if (GATHER) {
    int srow = RowIdx[ar];                       // invalid tail rows read row 0;
    aRow = (size_t)(srow >= 0 ? srow : 0) * K;   // their outputs are discarded
  } else {
    aRow = (size_t)(rowBase + ar) * K;
  }
  // B loader: thread t owns row t/2, k-window (t&1)*32 (4 chunks of 8 bf16)
  const int bn  = t >> 1;
  const int bkb = (t & 1) * 32;
  const size_t bRow = (size_t)(colBase + bn) * K;

  auto load_tile = [&](int kt, int bb) {
    lds_copy16_async(A + aRow + kt + akb,      &As[bb][ar][akb]);
    lds_copy16_async(A + aRow + kt + akb + 8,  &As[bb][ar][akb + 8]);
    #pragma unroll
    for (int j = 0; j < 4; ++j)
      lds_copy16_async(Bt + bRow + kt + bkb + j * 8, &Bs[bb][bn][bkb + j * 8]);
  };

  const int nt = K / KT;
  load_tile(0, 0);
  wait_async_lds();
  __syncthreads();

  const int m0 = waveRow * 32 + (lane & 15), m1 = m0 + 16;
  const int n0 = waveCol * 32 + (lane & 15), n1 = n0 + 16;
  const int khA = (lane < 16) ? 0 : 8;    // A 16-bit layout: K 0-7/16-23 vs 8-15/24-31
  const int khB = (lane < 16) ? 0 : 8;    // B 16-bit layout: K 0-7/16-23 vs 8-15/24-31

  for (int ti = 0; ti < nt; ++ti) {
    const int bb = ti & 1;
    if (ti + 1 < nt) load_tile((ti + 1) * KT, bb ^ 1);

    #pragma unroll
    for (int kk = 0; kk < KT; kk += 32) {
      union { bf16x16 v; uint4 q[2]; } a0, a1, b0, b1;
      const int ka = kk + khA, kb = kk + khB;
      a0.q[0] = *(const uint4*)&As[bb][m0][ka];
      a0.q[1] = *(const uint4*)&As[bb][m0][ka + 16];
      a1.q[0] = *(const uint4*)&As[bb][m1][ka];
      a1.q[1] = *(const uint4*)&As[bb][m1][ka + 16];
      b0.q[0] = *(const uint4*)&Bs[bb][n0][kb];
      b0.q[1] = *(const uint4*)&Bs[bb][n0][kb + 16];
      b1.q[0] = *(const uint4*)&Bs[bb][n1][kb];
      b1.q[1] = *(const uint4*)&Bs[bb][n1][kb + 16];

      acc00 = __builtin_amdgcn_wmma_f32_16x16x32_bf16(false, a0.v, false, b0.v, (short)0, acc00, false, false);
      acc01 = __builtin_amdgcn_wmma_f32_16x16x32_bf16(false, a0.v, false, b1.v, (short)0, acc01, false, false);
      acc10 = __builtin_amdgcn_wmma_f32_16x16x32_bf16(false, a1.v, false, b0.v, (short)0, acc10, false, false);
      acc11 = __builtin_amdgcn_wmma_f32_16x16x32_bf16(false, a1.v, false, b1.v, (short)0, acc11, false, false);
    }
    wait_async_lds();
    __syncthreads();
  }

  // Epilogue: bias + ReLU. C layout: VGPR i -> M=i (lanes 0-15) / i+8 (16-31), N=lane%16
  const int moff = (lane < 16) ? 0 : 8;
  f32x8 accs[2][2] = {{acc00, acc01}, {acc10, acc11}};
  #pragma unroll
  for (int tr = 0; tr < 2; ++tr) {
    #pragma unroll
    for (int tc = 0; tc < 2; ++tc) {
      const int col = colBase + waveCol * 32 + tc * 16 + (lane & 15);
      const float bv = bias[col];
      #pragma unroll
      for (int i = 0; i < 8; ++i) {
        const int m = waveRow * 32 + tr * 16 + moff + i;
        float v = accs[tr][tc][i] + bv;
        v = v > 0.f ? v : 0.f;
        if (GATHER) {
          int srow = RowIdx[m];
          if (srow >= 0) C[(size_t)srow * N + col] = v;
        } else {
          C[(size_t)(rowBase + m) * N + col] = v;
        }
      }
    }
  }
}

// ---------------------------------------------------------------------------
// Support kernels
// ---------------------------------------------------------------------------
__global__ void transpose_to_bf16(const float* __restrict__ in,
                                  unsigned short* __restrict__ out, int K, int N) {
  const size_t total = (size_t)K * N;
  const float* inp = in + (size_t)blockIdx.y * total;
  unsigned short* outp = out + (size_t)blockIdx.y * total;
  for (size_t i = blockIdx.x * (size_t)blockDim.x + threadIdx.x; i < total;
       i += (size_t)gridDim.x * blockDim.x) {
    int k = (int)(i / N), n = (int)(i % N);
    outp[(size_t)n * K + k] = f2bf_u(inp[i]);
  }
}

__global__ void f32_to_bf16(const float* __restrict__ in,
                            unsigned short* __restrict__ out, size_t n) {
  for (size_t i = blockIdx.x * (size_t)blockDim.x + threadIdx.x; i < n;
       i += (size_t)gridDim.x * blockDim.x)
    out[i] = f2bf_u(in[i]);
}

__global__ void zero_f32(float* __restrict__ p, int n) {
  int i = blockIdx.x * blockDim.x + threadIdx.x;
  if (i < n) p[i] = 0.f;
}

__global__ void col_stats(const float* __restrict__ h, float* __restrict__ sums,
                          int Bn, int Hn) {
  const int c = blockIdx.x * blockDim.x + threadIdx.x;       // column
  const int rowsPer = Bn / gridDim.y;
  const int r0 = blockIdx.y * rowsPer;
  float s = 0.f, s2 = 0.f;
  for (int r = r0; r < r0 + rowsPer; ++r) {
    float v = h[(size_t)r * Hn + c];
    s += v; s2 += v * v;
  }
  atomicAdd(&sums[c], s);
  atomicAdd(&sums[Hn + c], s2);
}

__global__ void bn_finalize(const float* __restrict__ sums,
                            const float* __restrict__ gamma,
                            const float* __restrict__ beta,
                            float* __restrict__ ss, int Bn, int Hn) {
  const int c = blockIdx.x * blockDim.x + threadIdx.x;
  if (c >= Hn) return;
  const float invB = 1.f / (float)Bn;
  const float mean = sums[c] * invB;
  const float var  = sums[Hn + c] * invB - mean * mean;
  const float sc   = gamma[c] * rsqrtf(var + 1e-5f);
  ss[c] = sc;
  ss[Hn + c] = beta[c] - mean * sc;
}

__global__ void bn_apply_bf16(const float* __restrict__ h,
                              const float* __restrict__ ss,
                              unsigned short* __restrict__ hb,
                              int Hn, size_t total) {
  for (size_t i = blockIdx.x * (size_t)blockDim.x + threadIdx.x; i < total;
       i += (size_t)gridDim.x * blockDim.x) {
    int c = (int)(i & (size_t)(Hn - 1));
    hb[i] = f2bf_u(h[i] * ss[c] + ss[Hn + c]);
  }
}

__global__ __launch_bounds__(256) void routing_kernel(
    const unsigned short* __restrict__ hb,
    const float* __restrict__ dqnW, const float* __restrict__ dqnb,
    int* __restrict__ actions, int* __restrict__ counts) {
  __shared__ float Wl[HDIM * NEXP];
  __shared__ float bl[NEXP];
  for (int i = threadIdx.x; i < HDIM * NEXP; i += blockDim.x) Wl[i] = dqnW[i];
  if (threadIdx.x < NEXP) bl[threadIdx.x] = dqnb[threadIdx.x];
  __syncthreads();
  const int s = blockIdx.x * blockDim.x + threadIdx.x;
  if (s >= B_SZ) return;
  float acc[NEXP];
  #pragma unroll
  for (int e = 0; e < NEXP; ++e) acc[e] = bl[e];
  const unsigned short* row = hb + (size_t)s * HDIM;
  for (int hh = 0; hh < HDIM; ++hh) {
    float a = bf2f(row[hh]);
    #pragma unroll
    for (int e = 0; e < NEXP; ++e) acc[e] += a * Wl[hh * NEXP + e];
  }
  int best = 0; float bv = acc[0];
  #pragma unroll
  for (int e = 1; e < NEXP; ++e)
    if (acc[e] > bv) { bv = acc[e]; best = e; }   // first-max like jnp.argmax
  actions[s] = best;
  atomicAdd(&counts[best], 1);
}

__global__ void scan_experts(const int* __restrict__ counts,
                             int* __restrict__ bases, int* __restrict__ cursors) {
  if (threadIdx.x == 0 && blockIdx.x == 0) {
    int acc = 0;
    for (int e = 0; e < NEXP; ++e) { bases[e] = acc; cursors[e] = acc; acc += counts[e]; }
  }
}

__global__ void scatter_idx(const int* __restrict__ actions,
                            int* __restrict__ cursors, int* __restrict__ idxlist, int Bn) {
  const int s = blockIdx.x * blockDim.x + threadIdx.x;
  if (s >= Bn) return;
  const int a = actions[s];
  const int pos = atomicAdd(&cursors[a], 1);
  idxlist[pos] = s;
}

// ---------------------------------------------------------------------------
extern "C" void kernel_launch(void* const* d_in, const int* in_sizes, int n_in,
                              void* d_out, int out_size, void* d_ws, size_t ws_size,
                              hipStream_t stream) {
  const float* x    = (const float*)d_in[0];
  const float* dqnW = (const float*)d_in[1];
  const float* dqnb = (const float*)d_in[2];
  const float* Ws   = (const float*)d_in[3];
  const float* bs   = (const float*)d_in[4];
  const float* g0   = (const float*)d_in[5];
  const float* b0   = (const float*)d_in[6];
  const float* We   = (const float*)d_in[7];
  const float* be   = (const float*)d_in[8];
  const float* gmid = (const float*)d_in[9];
  const float* bmid = (const float*)d_in[10];
  const float* Wend = (const float*)d_in[11];
  const float* bend = (const float*)d_in[12];
  float* out = (float*)d_out;

  char* ws = (char*)d_ws;
  size_t off = 0;
  auto alloc = [&](size_t bytes) -> void* {
    void* p = ws + off;
    off = (off + bytes + 255) & ~(size_t)255;
    return p;
  };
  unsigned short* xb    = (unsigned short*)alloc((size_t)B_SZ * DIN * 2);
  unsigned short* WsT   = (unsigned short*)alloc((size_t)HDIM * DIN * 2);
  unsigned short* WeT   = (unsigned short*)alloc((size_t)NMIDL * NEXP * HDIM * HDIM * 2);
  unsigned short* WendT = (unsigned short*)alloc((size_t)DOUTDIM * HDIM * 2);
  float*          h     = (float*)alloc((size_t)B_SZ * HDIM * 4);
  unsigned short* hb    = (unsigned short*)alloc((size_t)B_SZ * HDIM * 2);
  float*          sums  = (float*)alloc(2 * HDIM * 4);
  float*          ss    = (float*)alloc(2 * HDIM * 4);
  int*            actions = (int*)alloc(B_SZ * 4);
  int*            idxlist = (int*)alloc(B_SZ * 4);
  int*            counts  = (int*)alloc(NEXP * 4);
  int*            bases   = (int*)alloc(NEXP * 4);
  int*            cursors = (int*)alloc(NEXP * 4);

  // Weight + input conversion (bf16, weights transposed to [N,K])
  transpose_to_bf16<<<dim3(256, 1), 256, 0, stream>>>(Ws, WsT, DIN, HDIM);
  transpose_to_bf16<<<dim3(128, NMIDL * NEXP), 256, 0, stream>>>(We, WeT, HDIM, HDIM);
  transpose_to_bf16<<<dim3(64, 1), 256, 0, stream>>>(Wend, WendT, HDIM, DOUTDIM);
  f32_to_bf16<<<dim3(2048), 256, 0, stream>>>(x, xb, (size_t)B_SZ * DIN);

  // Start layer: h = relu(x @ Ws + bs)
  gemm_kernel<false><<<dim3(B_SZ / 64, HDIM / 128, 1), 256, 0, stream>>>(
      xb, WsT, bs, h, B_SZ, HDIM, DIN, nullptr, nullptr, nullptr, 0, 0);

  auto run_bn = [&](const float* gamma, const float* beta) {
    zero_f32<<<dim3(4), 256, 0, stream>>>(sums, 2 * HDIM);
    col_stats<<<dim3(HDIM / 256, 64), 256, 0, stream>>>(h, sums, B_SZ, HDIM);
    bn_finalize<<<dim3(2), 256, 0, stream>>>(sums, gamma, beta, ss, B_SZ, HDIM);
    bn_apply_bf16<<<dim3(2048), 256, 0, stream>>>(h, ss, hb, HDIM, (size_t)B_SZ * HDIM);
  };
  run_bn(g0, b0);

  for (int l = 0; l < NMIDL; ++l) {
    zero_f32<<<dim3(1), 32, 0, stream>>>((float*)counts, NEXP);
    routing_kernel<<<dim3(B_SZ / 256), 256, 0, stream>>>(hb, dqnW, dqnb, actions, counts);
    scan_experts<<<1, 1, 0, stream>>>(counts, bases, cursors);
    scatter_idx<<<dim3(B_SZ / 256), 256, 0, stream>>>(actions, cursors, idxlist, B_SZ);
    // Sparse expert dispatch: only the chosen expert per sample (8x fewer FLOPs)
    gemm_kernel<true><<<dim3(B_SZ / 64, HDIM / 128, NEXP), 256, 0, stream>>>(
        hb, WeT + (size_t)l * NEXP * HDIM * HDIM, be + (size_t)l * NEXP * HDIM, h,
        B_SZ, HDIM, HDIM, idxlist, counts, bases, HDIM * HDIM, HDIM);
    run_bn(gmid + (size_t)l * HDIM, bmid + (size_t)l * HDIM);
  }

  // End layer: out = relu(h @ Wend + bend)
  gemm_kernel<false><<<dim3(B_SZ / 64, DOUTDIM / 128, 1), 256, 0, stream>>>(
      hb, WendT, bend, out, B_SZ, DOUTDIM, HDIM, nullptr, nullptr, nullptr, 0, 0);
}
